// DeepStatisticalSolver_21509196219215
// MI455X (gfx1250) — matrix operand order repaired and checked by reading, not compiled, not run on hardware
//
#include <hip/hip_runtime.h>
#include <math.h>

#define N_NODES 100000
#define E_EDGES 800000
#define K_ITERS 3
#define ALPHA_C 0.1f
#define GAMMA_C 0.9f

typedef __attribute__((ext_vector_type(16))) _Float16 v16h;
typedef __attribute__((ext_vector_type(8)))  _Float16 h8;
typedef __attribute__((ext_vector_type(4)))  _Float16 h4;
typedef __attribute__((ext_vector_type(8)))  float    v8f;
typedef __attribute__((ext_vector_type(4)))  int      i4;

typedef __attribute__((address_space(1))) i4 gi4;   // global int4
typedef __attribute__((address_space(3))) i4 li4;   // LDS int4

#if __has_builtin(__builtin_amdgcn_global_load_async_to_lds_b128)
#define HAVE_ASYNC_LDS 1
#else
#define HAVE_ASYNC_LDS 0
#endif

// copy 16 bytes global -> LDS (async gather lane op when available)
static __device__ __forceinline__ void gather16(const _Float16* gp, _Float16* lp) {
#if HAVE_ASYNC_LDS
    gi4* g = (gi4*)(unsigned long long)gp;                    // AS1: full 64-bit VA
    li4* l = (li4*)(unsigned)(unsigned long long)lp;          // AS3: low 32 bits = LDS offset
    __builtin_amdgcn_global_load_async_to_lds_b128(g, l, 0, 0);
#else
    *(h8*)lp = *(const h8*)gp;
#endif
}
static __device__ __forceinline__ void wait_async() {
#if HAVE_ASYNC_LDS
#if __has_builtin(__builtin_amdgcn_s_wait_asynccnt)
    __builtin_amdgcn_s_wait_asynccnt(0);
#else
    asm volatile("s_wait_asynccnt 0x0" ::: "memory");
#endif
#endif
}

// A-matrix fragment, 16x32 f16 (ISA 7.12.2): row = lane&15; elems 0-7 hold
// K = kbase + hi*8 + i, elems 8-15 hold K = kbase + 16 + hi*8 + i  (hi = lane>=16).
// Both chunks are 16B-contiguous -> ds_load_b128 x2.
static __device__ __forceinline__ v16h lds_a_frag(const _Float16* row, int lane, int kbase) {
    int hi = (lane >> 4) & 1;
    h8 lo = *(const h8*)(row + kbase + hi * 8);
    h8 hh = *(const h8*)(row + kbase + 16 + hi * 8);
    return __builtin_shufflevector(lo, hh, 0, 1, 2, 3, 4, 5, 6, 7,
                                   8, 9, 10, 11, 12, 13, 14, 15);
}

static __device__ __forceinline__ v8f wmma_f16(v16h a, v16h b, v8f c) {
    return __builtin_amdgcn_wmma_f32_16x16x32_f16(false, a, false, b, (short)0, c, false, false);
}

// convert 8 contiguous f32 -> h8
static __device__ __forceinline__ h8 cvt8(const float* p) {
    const float4* q = (const float4*)p;
    float4 v0 = q[0], v1 = q[1];
    h8 r;
    r[0] = (_Float16)v0.x; r[1] = (_Float16)v0.y; r[2] = (_Float16)v0.z; r[3] = (_Float16)v0.w;
    r[4] = (_Float16)v1.x; r[5] = (_Float16)v1.y; r[6] = (_Float16)v1.z; r[7] = (_Float16)v1.w;
    return r;
}

// Build per-lane B fragments in LDS: frag[kb][lane][i] = W[kb*32 + (lane>>4)*8 + (i&7) + (i>>3)*16][lane&15]
// (f16, zero-padded for K >= kin). Each thread fills 4 consecutive halves.
static __device__ __forceinline__ void build_bfrag(const float* w, _Float16* frag,
                                                   int nelem, int kin, int tid) {
    for (int base = tid * 4; base < nelem; base += 256 * 4) {
        int ln = (base >> 4) & 31;
        int kb = base >> 9;
        int i0 = base & 15;
        int n  = ln & 15;
        h4 t;
#pragma unroll
        for (int j = 0; j < 4; ++j) {
            int k = kb * 32 + ((ln >> 4) << 3) + ((i0 + j) & 7) + (((i0 + j) >> 3) << 4);
            t[j] = (k < kin) ? (_Float16)w[k * 16 + n] : (_Float16)0.f;
        }
        *(h4*)&frag[base] = t;
    }
}

// ---------------------------------------------------------------------------
// Edge MLP + scatter: one 16-edge tile per wave.
// row = [h16[xi] (16) | h16[xj] (16) | edge_attr (3) | 0-pad -> 64]
// out[agg*16+n] += relu(x@W1+b1)@W2 + b2   (masked src!=dst)
// ---------------------------------------------------------------------------
__global__ __launch_bounds__(256) void edge_mlp_scatter(
    const _Float16* __restrict__ h16, const int* xi_idx, const int* xj_idx,
    const int* agg_idx, const int* src, const int* dst,
    const float* __restrict__ edge_attr,
    const float* __restrict__ w1, const float* __restrict__ b1,
    const float* __restrict__ w2, const float* __restrict__ b2,
    float* __restrict__ out, int nE)
{
    __shared__ __align__(16) _Float16 W1f[2 * 32 * 16];
    __shared__ __align__(16) _Float16 W2f[32 * 16];
    __shared__ float b1s[16], b2s[16];
    __shared__ __align__(16) _Float16 As[8][16 * 64];
    __shared__ __align__(16) _Float16 A2s[8][16 * 32];
    __shared__ __align__(16) int Agg[8][16];

    const int tid = threadIdx.x;
    build_bfrag(w1, W1f, 1024, 35, tid);
    build_bfrag(w2, W2f, 512, 16, tid);
    if (tid < 16) { b1s[tid] = b1[tid]; b2s[tid] = b2[tid]; }
    if (tid < 128) {                 // zero layer-2 K padding (cols 16..31) once
        int wv = tid >> 4, m = tid & 15;
        h8 z = {};
        *(h8*)&A2s[wv][m * 32 + 16] = z;
        *(h8*)&A2s[wv][m * 32 + 24] = z;
    }

    const int wave = tid >> 5, lane = tid & 31;
    const long tile = ((long)blockIdx.x * 8 + wave) * 16;

    // Gather h16 rows into A tile, 16B chunks: chunk g -> row r=g>>2, part c=g&3.
#pragma unroll
    for (int j = 0; j < 2; ++j) {
        int g = j * 32 + lane;
        int r = g >> 2, c = g & 3;
        long e = tile + r;
        long ec = (e < nE) ? e : 0;
        int idx = (c < 2) ? xi_idx[ec] : xj_idx[ec];
        gather16(h16 + (long)idx * 16 + (c & 1) * 8, &As[wave][r * 64 + c * 8]);
    }
    // edge_attr + tail padding + scatter metadata (one lane per edge row)
    {
        int r = lane >> 1, half = lane & 1;
        long e = tile + r;
        if (half == 1) {
            h8 at = {};
            if (e < nE) {
                at[0] = (_Float16)edge_attr[e * 3 + 0];
                at[1] = (_Float16)edge_attr[e * 3 + 1];
                at[2] = (_Float16)edge_attr[e * 3 + 2];
            }
            _Float16* Arow = &As[wave][r * 64];
            h8 z = {};
            *(h8*)&Arow[32] = at;
            *(h8*)&Arow[40] = z;
            *(h8*)&Arow[48] = z;
            *(h8*)&Arow[56] = z;
        } else {
            Agg[wave][r] = (e < nE && src[e] != dst[e]) ? agg_idx[e] : -1;
        }
    }
    wait_async();
    __syncthreads();

    const _Float16* myA = &As[wave][(lane & 15) * 64];
    v8f c1 = {};
    c1 = wmma_f16(lds_a_frag(myA, lane, 0),  *(const v16h*)&W1f[(lane) * 16],       c1);
    c1 = wmma_f16(lds_a_frag(myA, lane, 32), *(const v16h*)&W1f[512 + (lane) * 16], c1);

    const float bias1 = b1s[lane & 15];
    const int mofs = (lane >> 4) * 8;
    _Float16* A2w = A2s[wave];
#pragma unroll
    for (int q = 0; q < 8; ++q) {
        float v = c1[q] + bias1;
        v = v > 0.f ? v : 0.f;
        A2w[(q + mofs) * 32 + (lane & 15)] = (_Float16)v;   // C layout -> A layout
    }
    __syncthreads();

    const _Float16* myA2 = &A2s[wave][(lane & 15) * 32];
    v8f c2 = {};
    c2 = wmma_f16(lds_a_frag(myA2, lane, 0), *(const v16h*)&W2f[lane * 16], c2);

    const float bias2 = b2s[lane & 15];
    int ag[8];
    *(int4*)&ag[0] = *(const int4*)&Agg[wave][mofs];
    *(int4*)&ag[4] = *(const int4*)&Agg[wave][mofs + 4];
#pragma unroll
    for (int q = 0; q < 8; ++q) {
        if (ag[q] >= 0)
            atomicAdd(&out[(long)ag[q] * 16 + (lane & 15)], c2[q] + bias2);
    }
}

// ---------------------------------------------------------------------------
// Node psi MLP: h += ALPHA * mlp2([h | mess_to | mess_from | prb]); h16 mirror.
// ---------------------------------------------------------------------------
__global__ __launch_bounds__(256) void node_psi(
    float* __restrict__ h, _Float16* __restrict__ h16,
    const float* __restrict__ mt, const float* __restrict__ mf,
    const float* __restrict__ prb,
    const float* __restrict__ w1, const float* __restrict__ b1,
    const float* __restrict__ w2, const float* __restrict__ b2, int nN)
{
    __shared__ __align__(16) _Float16 W1f[2 * 32 * 16];
    __shared__ __align__(16) _Float16 W2f[32 * 16];
    __shared__ float b1s[16], b2s[16];
    __shared__ __align__(16) _Float16 As[8][16 * 64];
    __shared__ __align__(16) _Float16 A2s[8][16 * 32];

    const int tid = threadIdx.x;
    build_bfrag(w1, W1f, 1024, 49, tid);
    build_bfrag(w2, W2f, 512, 16, tid);
    if (tid < 16) { b1s[tid] = b1[tid]; b2s[tid] = b2[tid]; }
    if (tid < 128) {
        int wv = tid >> 4, m = tid & 15;
        h8 z = {};
        *(h8*)&A2s[wv][m * 32 + 16] = z;
        *(h8*)&A2s[wv][m * 32 + 24] = z;
    }

    const int wave = tid >> 5, lane = tid & 31;
    const long tile = ((long)blockIdx.x * 8 + wave) * 16;

    {   // fill: lane pair per node row (contiguous, f32 -> f16)
        int r = lane >> 1, half = lane & 1;
        long nd = tile + r;
        long nc = (nd < nN) ? nd : 0;
        _Float16* Arow = &As[wave][r * 64];
        if (half == 0) {
            *(h8*)&Arow[0]  = cvt8(h  + nc * 16);
            *(h8*)&Arow[8]  = cvt8(h  + nc * 16 + 8);
            *(h8*)&Arow[16] = cvt8(mt + nc * 16);
            *(h8*)&Arow[24] = cvt8(mt + nc * 16 + 8);
        } else {
            *(h8*)&Arow[32] = cvt8(mf + nc * 16);
            *(h8*)&Arow[40] = cvt8(mf + nc * 16 + 8);
            h8 pz = {};
            pz[0] = (_Float16)prb[nc];
            *(h8*)&Arow[48] = pz;
            h8 z = {};
            *(h8*)&Arow[56] = z;
        }
    }
    __syncthreads();

    const _Float16* myA = &As[wave][(lane & 15) * 64];
    v8f c1 = {};
    c1 = wmma_f16(lds_a_frag(myA, lane, 0),  *(const v16h*)&W1f[lane * 16],       c1);
    c1 = wmma_f16(lds_a_frag(myA, lane, 32), *(const v16h*)&W1f[512 + lane * 16], c1);

    const float bias1 = b1s[lane & 15];
    const int mofs = (lane >> 4) * 8;
    _Float16* A2w = A2s[wave];
#pragma unroll
    for (int q = 0; q < 8; ++q) {
        float v = c1[q] + bias1;
        v = v > 0.f ? v : 0.f;
        A2w[(q + mofs) * 32 + (lane & 15)] = (_Float16)v;
    }
    __syncthreads();

    const _Float16* myA2 = &A2s[wave][(lane & 15) * 32];
    v8f c2 = {};
    c2 = wmma_f16(lds_a_frag(myA2, lane, 0), *(const v16h*)&W2f[lane * 16], c2);

    const float bias2 = b2s[lane & 15];
#pragma unroll
    for (int q = 0; q < 8; ++q) {
        long nd = tile + q + mofs;
        if (nd < nN) {
            long ix = nd * 16 + (lane & 15);
            float hv = h[ix] + ALPHA_C * (c2[q] + bias2);
            h[ix] = hv;
            h16[ix] = (_Float16)hv;   // f16 mirror for edge gathers
        }
    }
}

// ---------------------------------------------------------------------------
// Decoder: u = relu(h@W1+b1)@w2 + b2  (16->16->1, negligible FLOPs)
// ---------------------------------------------------------------------------
__global__ void dec_kernel(const float* __restrict__ h,
                           const float* __restrict__ w1, const float* __restrict__ b1,
                           const float* __restrict__ w2, const float* __restrict__ b2,
                           float* __restrict__ u, int nN)
{
    int i = blockIdx.x * blockDim.x + threadIdx.x;
    if (i >= nN) return;
    const float* hr = h + (long)i * 16;
    float acc = b2[0];
#pragma unroll
    for (int j = 0; j < 16; ++j) {
        float z = b1[j];
#pragma unroll
        for (int k = 0; k < 16; ++k) z += hr[k] * w1[k * 16 + j];
        z = z > 0.f ? z : 0.f;
        acc += z * w2[j];
    }
    u[i] = acc;
}

__global__ void residual_kernel(const int* src, const int* dst,
                                const float* __restrict__ a,
                                const float* __restrict__ u,
                                float* __restrict__ Au, int nE)
{
    int e = blockIdx.x * blockDim.x + threadIdx.x;
    if (e < nE) atomicAdd(&Au[src[e]], a[e] * u[dst[e]]);
}

__global__ void loss_kernel(const float* __restrict__ Au, const float* __restrict__ y,
                            float* __restrict__ out_loss, float wgt, int n)
{
    __shared__ float red[256];
    float acc = 0.f;
    for (int i = blockIdx.x * blockDim.x + threadIdx.x; i < n;
         i += gridDim.x * blockDim.x) {
        float d = Au[i] - y[i];
        acc += d * d;
    }
    red[threadIdx.x] = acc;
    __syncthreads();
    for (int s = 128; s > 0; s >>= 1) {
        if (threadIdx.x < s) red[threadIdx.x] += red[threadIdx.x + s];
        __syncthreads();
    }
    if (threadIdx.x == 0) atomicAdd(out_loss, red[0] * wgt);
}

__global__ void zero_kernel(float* p, long n)
{
    long i = (long)blockIdx.x * blockDim.x + threadIdx.x;
    if (i < n) p[i] = 0.f;
}

// ---------------------------------------------------------------------------
extern "C" void kernel_launch(void* const* d_in, const int* in_sizes, int n_in,
                              void* d_out, int out_size, void* d_ws, size_t ws_size,
                              hipStream_t stream)
{
    const int*   edge_index = (const int*)d_in[0];
    const float* edge_attr  = (const float*)d_in[1];
    const float* a_ij       = (const float*)d_in[2];
    const float* prb        = (const float*)d_in[3];
    // d_in[4] = sol (unused)
    const float* y          = (const float*)d_in[5];
    const float* phi_to_w1   = (const float*)d_in[6];
    const float* phi_to_b1   = (const float*)d_in[7];
    const float* phi_to_w2   = (const float*)d_in[8];
    const float* phi_to_b2   = (const float*)d_in[9];
    const float* phi_from_w1 = (const float*)d_in[10];
    const float* phi_from_b1 = (const float*)d_in[11];
    const float* phi_from_w2 = (const float*)d_in[12];
    const float* phi_from_b2 = (const float*)d_in[13];
    const float* psi_w1      = (const float*)d_in[14];
    const float* psi_b1      = (const float*)d_in[15];
    const float* psi_w2      = (const float*)d_in[16];
    const float* psi_b2      = (const float*)d_in[17];
    const float* dec_w1      = (const float*)d_in[18];
    const float* dec_b1      = (const float*)d_in[19];
    const float* dec_w2      = (const float*)d_in[20];
    const float* dec_b2      = (const float*)d_in[21];

    const int* src = edge_index;
    const int* dst = edge_index + E_EDGES;

    float* ws = (float*)d_ws;
    float* h    = ws;                              // N*16 f32
    float* mt   = ws + (long)N_NODES * 16;         // N*16 f32
    float* mf   = ws + (long)N_NODES * 32;         // N*16 f32
    float* Au   = ws + (long)N_NODES * 48;         // N   f32
    _Float16* h16 = (_Float16*)(ws + (long)N_NODES * 49);  // N*16 f16

    float* u_out    = (float*)d_out;
    float* loss_out = (float*)d_out + N_NODES;

    const int edge_blocks = (E_EDGES + 127) / 128;
    const int node_blocks = (N_NODES + 127) / 128;

    // init: h = 0, h16 = 0, loss = 0  (h16 zeroed as N*8 f32 words)
    zero_kernel<<<((long)N_NODES * 16 + 255) / 256, 256, 0, stream>>>(h, (long)N_NODES * 16);
    zero_kernel<<<((long)N_NODES * 8 + 255) / 256, 256, 0, stream>>>((float*)h16, (long)N_NODES * 8);
    zero_kernel<<<1, 256, 0, stream>>>(loss_out, 1);

    for (int t = 0; t < K_ITERS; ++t) {
        const float* pt_w1 = phi_to_w1   + (long)t * 35 * 16;
        const float* pt_b1 = phi_to_b1   + (long)t * 16;
        const float* pt_w2 = phi_to_w2   + (long)t * 16 * 16;
        const float* pt_b2 = phi_to_b2   + (long)t * 16;
        const float* pf_w1 = phi_from_w1 + (long)t * 35 * 16;
        const float* pf_b1 = phi_from_b1 + (long)t * 16;
        const float* pf_w2 = phi_from_w2 + (long)t * 16 * 16;
        const float* pf_b2 = phi_from_b2 + (long)t * 16;
        const float* ps_w1 = psi_w1 + (long)t * 49 * 16;
        const float* ps_b1 = psi_b1 + (long)t * 16;
        const float* ps_w2 = psi_w2 + (long)t * 16 * 16;
        const float* ps_b2 = psi_b2 + (long)t * 16;
        const float* dc_w1 = dec_w1 + (long)t * 16 * 16;
        const float* dc_b1 = dec_b1 + (long)t * 16;
        const float* dc_w2 = dec_w2 + (long)t * 16;
        const float* dc_b2 = dec_b2 + (long)t;

        zero_kernel<<<((long)N_NODES * 32 + 255) / 256, 256, 0, stream>>>(mt, (long)N_NODES * 32);

        // mess_to: xi=dst, xj=src, agg=dst
        edge_mlp_scatter<<<edge_blocks, 256, 0, stream>>>(
            h16, dst, src, dst, src, dst, edge_attr,
            pt_w1, pt_b1, pt_w2, pt_b2, mt, E_EDGES);
        // mess_from: xi=src, xj=dst, agg=src
        edge_mlp_scatter<<<edge_blocks, 256, 0, stream>>>(
            h16, src, dst, src, src, dst, edge_attr,
            pf_w1, pf_b1, pf_w2, pf_b2, mf, E_EDGES);

        node_psi<<<node_blocks, 256, 0, stream>>>(
            h, h16, mt, mf, prb, ps_w1, ps_b1, ps_w2, ps_b2, N_NODES);

        dec_kernel<<<(N_NODES + 255) / 256, 256, 0, stream>>>(
            h, dc_w1, dc_b1, dc_w2, dc_b2, u_out, N_NODES);

        zero_kernel<<<(N_NODES + 255) / 256, 256, 0, stream>>>(Au, N_NODES);
        residual_kernel<<<(E_EDGES + 255) / 256, 256, 0, stream>>>(
            src, dst, a_ij, u_out, Au, E_EDGES);

        float wgt = powf(GAMMA_C, (float)(K_ITERS - t - 1)) / (float)N_NODES;
        loss_kernel<<<512, 256, 0, stream>>>(Au, y, loss_out, wgt, N_NODES);
    }
}
